// CrissCrossAttentionDecoder_7000796692592
// MI455X (gfx1250) — matrix operand order repaired and checked by reading, not compiled, not run on hardware
//
#include <hip/hip_runtime.h>
#include <cstdint>
#include <cstddef>

// ---------------- problem constants ----------------
#define B_   4
#define C_   512
#define CQ_  64
#define H_   160
#define W_   160
#define P_   (H_ * W_)        // 25600
#define L_   (W_ + H_ - 1)    // 319

// ---------------- WMMA types ----------------
typedef __attribute__((ext_vector_type(16))) __bf16 v16bf;
typedef __attribute__((ext_vector_type(8)))  float  v8f;

union U16x16 { v16bf v; uint4 q[2]; unsigned short s[16]; };
union U16x8  { uint4 q; unsigned short s[8]; };

__device__ __forceinline__ unsigned short f2bf(float f) {
  union { float f; unsigned u; } v; v.f = f;
  unsigned u = v.u;
  return (unsigned short)((u + 0x7FFFu + ((u >> 16) & 1u)) >> 16);
}

// A-fragment (16x32 bf16, M=lane&15): elems 0..7 = K kc+hi*8.., elems 8..15 = K kc+16+hi*8..
__device__ __forceinline__ v16bf load_a_frag(const unsigned short* row, int kc, int hi) {
  U16x16 u;
  u.q[0] = *reinterpret_cast<const uint4*>(row + kc + hi * 8);
  u.q[1] = *reinterpret_cast<const uint4*>(row + kc + 16 + hi * 8);
  return u.v;
}
// B-fragment (32x16 bf16, N=lane&15): elems 0..7 = K kc+hi*8.., elems 8..15 = K kc+16+hi*8..
__device__ __forceinline__ v16bf load_b_frag(const unsigned short* row, int kc, int hi) {
  U16x16 u;
  u.q[0] = *reinterpret_cast<const uint4*>(row + kc + hi * 8);
  u.q[1] = *reinterpret_cast<const uint4*>(row + kc + 16 + hi * 8);
  return u.v;
}

__device__ __forceinline__ v8f wmma_bf16(v16bf a, v16bf b, v8f c) {
  return __builtin_amdgcn_wmma_f32_16x16x32_bf16(false, a, false, b, (short)0, c, false, false);
}

// ---------- CDNA5 async global->LDS copy (guarded; falls back to ld+ds_store) ----------
#if defined(__has_builtin)
#if __has_builtin(__builtin_amdgcn_global_load_async_to_lds_b128)
#define HAS_ASYNC_LDS 1
#endif
#endif

typedef int vi4 __attribute__((vector_size(4 * sizeof(int))));
typedef __attribute__((address_space(1))) vi4 gvi4;   // global int4
typedef __attribute__((address_space(3))) vi4 lvi4;   // LDS int4

__device__ __forceinline__ void copy_b128_to_lds(const unsigned short* g, unsigned short* l) {
#ifdef HAS_ASYNC_LDS
  __builtin_amdgcn_global_load_async_to_lds_b128((gvi4*)(g), (lvi4*)(l), 0, 0);
#else
  *reinterpret_cast<uint4*>(l) = *reinterpret_cast<const uint4*>(g);
#endif
}
__device__ __forceinline__ void wait_lds_copies() {
#ifdef HAS_ASYNC_LDS
#if __has_builtin(__builtin_amdgcn_s_wait_asynccnt)
  __builtin_amdgcn_s_wait_asynccnt(0);
#else
  asm volatile("s_wait_asynccnt 0" ::: "memory");
#endif
#endif
}

// ---------------- workspace layout (bytes) ----------------
// Region A (reused): Qt/Mt live during projections; energy/att live afterwards.
static const size_t SZ_XT    = (size_t)B_ * P_ * C_ * 2;        // 104,857,600
static const size_t OFF_QT   = 0;
static const size_t OFF_MT   = SZ_XT;
static const size_t REGA_SZ  = 2 * SZ_XT;                       // 209,715,200
static const size_t OFF_EN   = 0;                               // f32 (B,P,319)
static const size_t SZ_EN    = (size_t)B_ * P_ * L_ * 4;        // 130,662,400
static const size_t OFF_AR   = OFF_EN + SZ_EN;                  // bf16 (B,H,W,160)
static const size_t SZ_AR    = (size_t)B_ * P_ * W_ * 2;        // 32,768,000
static const size_t OFF_AC   = OFF_AR + SZ_AR;                  // bf16 (B,W,H,160)
// Region B (persistent)
static const size_t REGB     = REGA_SZ;
static const size_t OFF_WQ16 = REGB;                            // 65,536
static const size_t OFF_WK16 = REGB + 65536;
static const size_t OFF_WV16 = REGB + 131072;                   // 524,288
static const size_t OFF_TS   = REGB + 655360;                   // bf16 (B,H,W,64)
static const size_t OFF_FS   = REGB + 13762560;
static const size_t OFF_V16  = REGB + 26869760;                 // bf16 (B,C,H,W)
static const size_t OFF_OBUF = REGB + 131727360;                // f32 (B,C,H,W)
static const size_t OFF_V16T = REGB + 341442560;                // bf16 (B,C,W,H)
// total required ws: REGB + 341,442,560 + 104,857,600 = 656,015,360 bytes (~626 MB)

// ---------------- kernels ----------------

// LDS-tiled transpose+convert: (B,C,P) f32 -> (B,P,C) bf16. grid (P/32, C/32, B), 256 thr
__global__ __launch_bounds__(256) void k_transpose_bf16(const float* __restrict__ in,
                                                        unsigned short* __restrict__ out) {
  __shared__ float tile[32][33];
  const int n = blockIdx.z;
  const int p0 = blockIdx.x * 32, c0 = blockIdx.y * 32;
  const int tx = threadIdx.x & 31, ty = threadIdx.x >> 5;  // ty 0..7
  const float* src = in + ((size_t)n * C_ + c0) * P_ + p0;
#pragma unroll
  for (int s = 0; s < 32; s += 8)
    tile[ty + s][tx] = src[(size_t)(ty + s) * P_ + tx];    // coalesced over p
  __syncthreads();
  unsigned short* dst = out + ((size_t)n * P_ + p0) * C_ + c0;
#pragma unroll
  for (int s = 0; s < 32; s += 8)
    dst[(size_t)(ty + s) * C_ + tx] = f2bf(tile[tx][ty + s]);  // coalesced over c
}

__global__ __launch_bounds__(256) void k_convert_bf16(const float* __restrict__ in,
                                                      unsigned short* __restrict__ out) {
  size_t idx = (size_t)blockIdx.x * 256 + threadIdx.x;
  out[idx] = f2bf(in[idx]);
}

// D[o,p] = (sum_c W16[o,c]*Xt[p,c] + bias[o]) * scale
// mode 0: out (B,P,O) bf16 ; mode 1: out (B,O,P) bf16 AND out2 (B,O,W,H) bf16
__global__ __launch_bounds__(256) void k_proj(const unsigned short* __restrict__ W16,
                                              const float* __restrict__ bias,
                                              const unsigned short* __restrict__ Xt,
                                              unsigned short* __restrict__ out,
                                              unsigned short* __restrict__ out2,
                                              int O, float scale, int mode) {
  const int tid = threadIdx.x;
  const int wave = tid >> 5, lane15 = tid & 15, hi = (tid >> 4) & 1;
  const int n  = blockIdx.z;
  const int o0 = blockIdx.y * 16;
  const int p0 = blockIdx.x * 128 + wave * 16;
  const int p  = p0 + lane15;

  const unsigned short* arow = W16 + (size_t)(o0 + lane15) * C_;
  const unsigned short* brow = Xt + ((size_t)n * P_ + p) * C_;

  v8f acc = {};
#pragma unroll
  for (int kc = 0; kc < C_; kc += 32)
    acc = wmma_bf16(load_a_frag(arow, kc, hi), load_b_frag(brow, kc, hi), acc);

  if (mode == 0) {
    U16x8 w;
#pragma unroll
    for (int r = 0; r < 8; ++r)
      w.s[r] = f2bf((acc[r] + bias[o0 + hi * 8 + r]) * scale);
    *reinterpret_cast<uint4*>(out + ((size_t)n * P_ + p) * O + o0 + hi * 8) = w.q;
  } else {
    const int h = p / W_, w = p % W_;
#pragma unroll
    for (int r = 0; r < 8; ++r) {
      int o = o0 + hi * 8 + r;
      unsigned short bf = f2bf((acc[r] + bias[o]) * scale);
      out [((size_t)n * O + o) * P_ + p] = bf;                      // (B,C,H,W)
      out2[((size_t)(n * O + o) * W_ + w) * H_ + h] = bf;           // (B,C,W,H)
    }
  }
}

// e_row[n,j,h,w] = sum_o t[o,h,w] f[o,h,j]  -> energy[(n,h,w), j]   (LDS-staged slabs)
__global__ __launch_bounds__(256) void k_erow(const unsigned short* __restrict__ ts,
                                              const unsigned short* __restrict__ fs,
                                              float* __restrict__ energy) {
  __shared__ __align__(16) unsigned short sf[W_ * CQ_];  // 20KB
  __shared__ __align__(16) unsigned short st[W_ * CQ_];  // 20KB
  const int tid = threadIdx.x;
  const int wave = tid >> 5, lane15 = tid & 15, hi = (tid >> 4) & 1;
  const int n = blockIdx.x / H_, h = blockIdx.x % H_;
  const unsigned short* fb = fs + (size_t)(n * H_ + h) * W_ * CQ_;
  const unsigned short* tb = ts + (size_t)(n * H_ + h) * W_ * CQ_;

  for (int i = tid; i < (W_ * CQ_) / 8; i += 256) {      // 1280 b128 copies per slab
    copy_b128_to_lds(fb + i * 8, sf + i * 8);
    copy_b128_to_lds(tb + i * 8, st + i * 8);
  }
  wait_lds_copies();
  __syncthreads();

  float* eb = energy + (size_t)(n * H_ + h) * W_ * L_;
  for (int t = wave; t < 100; t += 8) {
    int j0 = (t / 10) * 16, w0 = (t % 10) * 16;
    const unsigned short* arow = sf + (size_t)(j0 + lane15) * CQ_;   // A[m=j, k=o]
    const unsigned short* brow = st + (size_t)(w0 + lane15) * CQ_;   // B[k=o, n=w]
    v8f acc = {};
#pragma unroll
    for (int kc = 0; kc < CQ_; kc += 32)
      acc = wmma_bf16(load_a_frag(arow, kc, hi), load_b_frag(brow, kc, hi), acc);
    float* dst = eb + (size_t)(w0 + lane15) * L_;
#pragma unroll
    for (int r = 0; r < 8; ++r) dst[j0 + hi * 8 + r] = acc[r];
  }
}

// e_col[n,i,h,w] = sum_o t[o,h,w] f[o,i,w]  -> energy[(n,h,w), 160 + (i<h?i:i-1)], i!=h
__global__ __launch_bounds__(256) void k_ecol(const unsigned short* __restrict__ ts,
                                              const unsigned short* __restrict__ fs,
                                              float* __restrict__ energy) {
  __shared__ __align__(16) unsigned short sf[H_ * CQ_];
  __shared__ __align__(16) unsigned short st[H_ * CQ_];
  const int tid = threadIdx.x;
  const int wave = tid >> 5, lane15 = tid & 15, hi = (tid >> 4) & 1;
  const int n = blockIdx.x / W_, w = blockIdx.x % W_;

  for (int i = tid; i < (H_ * CQ_) / 8; i += 256) {
    int row = i >> 3, part = (i & 7) * 8;
    const unsigned short* fsrc = fs + ((size_t)(n * H_ + row) * W_ + w) * CQ_ + part;
    const unsigned short* tsrc = ts + ((size_t)(n * H_ + row) * W_ + w) * CQ_ + part;
    copy_b128_to_lds(fsrc, sf + row * CQ_ + part);
    copy_b128_to_lds(tsrc, st + row * CQ_ + part);
  }
  wait_lds_copies();
  __syncthreads();

  for (int t = wave; t < 100; t += 8) {
    int i0 = (t / 10) * 16, h0 = (t % 10) * 16;
    const unsigned short* arow = sf + (size_t)(i0 + lane15) * CQ_;   // A[m=i, k=o]
    const unsigned short* brow = st + (size_t)(h0 + lane15) * CQ_;   // B[k=o, n=h]
    v8f acc = {};
#pragma unroll
    for (int kc = 0; kc < CQ_; kc += 32)
      acc = wmma_bf16(load_a_frag(arow, kc, hi), load_b_frag(brow, kc, hi), acc);
    int h = h0 + lane15;
    float* dst = energy + ((size_t)(n * H_ + h) * W_ + w) * L_;
#pragma unroll
    for (int r = 0; r < 8; ++r) {
      int i = i0 + hi * 8 + r;
      if (i != h) { int k = (i < h) ? i : i - 1; dst[160 + k] = acc[r]; }
    }
  }
}

// softmax over 319; emit att_row_t (B,H,W,160) and diag-zero expanded att_col_t (B,W,H,160)
__global__ __launch_bounds__(256) void k_softmax(const float* __restrict__ energy,
                                                 unsigned short* __restrict__ arow,
                                                 unsigned short* __restrict__ acol) {
  size_t gid = (size_t)blockIdx.x * 256 + threadIdx.x;   // over B*P
  int n = (int)(gid / P_);
  int p = (int)(gid % P_);
  int h = p / W_, w = p % W_;
  const float* e = energy + gid * L_;

  float mx = -3.4e38f;
  for (int l = 0; l < L_; ++l) mx = fmaxf(mx, e[l]);
  float s = 0.f;
  for (int l = 0; l < L_; ++l) s += __expf(e[l] - mx);
  float inv = 1.f / s;

  unsigned short* ar = arow + gid * W_;
  for (int j = 0; j < W_; ++j) ar[j] = f2bf(__expf(e[j] - mx) * inv);

  unsigned short* ac = acol + ((size_t)(n * W_ + w) * H_ + h) * H_;
  ac[h] = 0;
  for (int k = 0; k < H_ - 1; ++k) {
    int i = (k < h) ? k : k + 1;
    ac[i] = f2bf(__expf(e[160 + k] - mx) * inv);
  }
}

// out_row[c,w] = sum_j v[c,h,j] * att_row_t[h,w,j]  -> Obuf (B,C,H,W) f32
__global__ __launch_bounds__(256) void k_rowagg(const unsigned short* __restrict__ v16,
                                                const unsigned short* __restrict__ arow,
                                                float* __restrict__ obuf) {
  const int tid = threadIdx.x;
  const int wave = tid >> 5, lane15 = tid & 15, hi = (tid >> 4) & 1;
  const int z = blockIdx.z, n = z / H_, h = z % H_;
  const int c0 = (blockIdx.y * 8 + wave) * 16;
  const int w0 = blockIdx.x * 16;

  const unsigned short* arowp = v16 + ((size_t)(n * C_ + c0 + lane15) * H_ + h) * W_;  // A[m=c,k=j]
  const unsigned short* brow  = arow + ((size_t)(n * H_ + h) * W_ + w0 + lane15) * W_; // B[k=j,n=w]
  v8f acc = {};
#pragma unroll
  for (int kc = 0; kc < W_; kc += 32)
    acc = wmma_bf16(load_a_frag(arowp, kc, hi), load_b_frag(brow, kc, hi), acc);
#pragma unroll
  for (int r = 0; r < 8; ++r) {
    int c = c0 + hi * 8 + r;
    obuf[((size_t)(n * C_ + c) * H_ + h) * W_ + w0 + lane15] = acc[r];
  }
}

// out_col[c,h] = sum_i v16t[c,w,i] * att_col_t[w,h,i]; final: out = gamma*(Obuf+col)+memory
__global__ __launch_bounds__(256) void k_colagg(const unsigned short* __restrict__ v16t,
                                                const unsigned short* __restrict__ acol,
                                                const float* __restrict__ obuf,
                                                const float* __restrict__ mem,
                                                const float* __restrict__ gamma,
                                                float* __restrict__ out) {
  const int tid = threadIdx.x;
  const int wave = tid >> 5, lane15 = tid & 15, hi = (tid >> 4) & 1;
  const int z = blockIdx.z, n = z / W_, w = z % W_;
  const int c0 = (blockIdx.y * 8 + wave) * 16;
  const int h0 = blockIdx.x * 16;

  const unsigned short* arowp = v16t + ((size_t)(n * C_ + c0 + lane15) * W_ + w) * H_;  // A[m=c,k=i]
  const unsigned short* brow  = acol + ((size_t)(n * W_ + w) * H_ + h0 + lane15) * H_;  // B[k=i,n=h]
  v8f acc = {};
#pragma unroll
  for (int kc = 0; kc < H_; kc += 32)
    acc = wmma_bf16(load_a_frag(arowp, kc, hi), load_b_frag(brow, kc, hi), acc);

  float g = gamma[0];
#pragma unroll
  for (int r = 0; r < 8; ++r) {
    int c = c0 + hi * 8 + r;
    size_t idx = ((size_t)(n * C_ + c) * H_ + h0 + lane15) * W_ + w;
    out[idx] = g * (obuf[idx] + acc[r]) + mem[idx];
  }
}

// ---------------- launch ----------------
extern "C" void kernel_launch(void* const* d_in, const int* in_sizes, int n_in,
                              void* d_out, int out_size, void* d_ws, size_t ws_size,
                              hipStream_t stream) {
  const float* query  = (const float*)d_in[0];
  const float* memory = (const float*)d_in[1];
  const float* Wq     = (const float*)d_in[2];
  const float* bq     = (const float*)d_in[3];
  const float* Wk     = (const float*)d_in[4];
  const float* bk     = (const float*)d_in[5];
  const float* Wv     = (const float*)d_in[6];
  const float* bv     = (const float*)d_in[7];
  const float* gamma  = (const float*)d_in[8];
  float* out = (float*)d_out;

  char* ws = (char*)d_ws;
  unsigned short* QT   = (unsigned short*)(ws + OFF_QT);
  unsigned short* MT   = (unsigned short*)(ws + OFF_MT);
  float*          EN   = (float*)(ws + OFF_EN);
  unsigned short* AR   = (unsigned short*)(ws + OFF_AR);
  unsigned short* AC   = (unsigned short*)(ws + OFF_AC);
  unsigned short* WQ16 = (unsigned short*)(ws + OFF_WQ16);
  unsigned short* WK16 = (unsigned short*)(ws + OFF_WK16);
  unsigned short* WV16 = (unsigned short*)(ws + OFF_WV16);
  unsigned short* TS   = (unsigned short*)(ws + OFF_TS);
  unsigned short* FS   = (unsigned short*)(ws + OFF_FS);
  unsigned short* V16  = (unsigned short*)(ws + OFF_V16);
  float*          OB   = (float*)(ws + OFF_OBUF);
  unsigned short* V16T = (unsigned short*)(ws + OFF_V16T);

  const float scaling = 0.125f;  // CQ^-0.5

  // 1) LDS-tiled transpose to (B,P,C) bf16 + bf16 weights
  k_transpose_bf16<<<dim3(P_ / 32, C_ / 32, B_), 256, 0, stream>>>(query, QT);
  k_transpose_bf16<<<dim3(P_ / 32, C_ / 32, B_), 256, 0, stream>>>(memory, MT);
  k_convert_bf16<<<(CQ_ * C_) / 256, 256, 0, stream>>>(Wq, WQ16);
  k_convert_bf16<<<(CQ_ * C_) / 256, 256, 0, stream>>>(Wk, WK16);
  k_convert_bf16<<<(C_ * C_) / 256, 256, 0, stream>>>(Wv, WV16);

  // 2) projections (WMMA GEMMs); V emitted in both (B,C,H,W) and (B,C,W,H) layouts
  k_proj<<<dim3(P_ / 128, CQ_ / 16, B_), 256, 0, stream>>>(WQ16, bq, QT, TS, nullptr, CQ_, scaling, 0);
  k_proj<<<dim3(P_ / 128, CQ_ / 16, B_), 256, 0, stream>>>(WK16, bk, MT, FS, nullptr, CQ_, 1.0f, 0);
  k_proj<<<dim3(P_ / 128, C_ / 16,  B_), 256, 0, stream>>>(WV16, bv, MT, V16, V16T, C_, 1.0f, 1);

  // 3) energies (WMMA, LDS-staged) — region A now reused for energy/att
  k_erow<<<B_ * H_, 256, 0, stream>>>(TS, FS, EN);
  k_ecol<<<B_ * W_, 256, 0, stream>>>(TS, FS, EN);

  // 4) softmax over 319
  k_softmax<<<(B_ * P_) / 256, 256, 0, stream>>>(EN, AR, AC);

  // 5) aggregation (WMMA) + fused epilogue
  k_rowagg<<<dim3(W_ / 16, C_ / 128, B_ * H_), 256, 0, stream>>>(V16, AR, OB);
  k_colagg<<<dim3(H_ / 16, C_ / 128, B_ * W_), 256, 0, stream>>>(V16T, AC, OB, memory, gamma, out);
}
